// SparseLocalFrameLayer_18399639896488
// MI455X (gfx1250) — compile-verified, matrix operand order checked by reading, not run on hardware
//
#include <hip/hip_runtime.h>
#include <hip/hip_bf16.h>

// ---------------- problem constants (from reference) ----------------
#define N_NODES    50000
#define N_EDGES    1600000
#define SCALAR_DIM 64
#define VECTOR_DIM 32
#define IN_DIM     163      // 2*64 + 32 + 3
#define HID        96
#define TILE       16       // edges per wave = WMMA M
#define WAVES      4        // waves per block (128 threads)

// LDS row paddings chosen for 16B alignment + bank spread
#define A_PAD  200          // halfs per A / W1T row  (400 B, 16B-aligned, gcd(100dw,64)=4)
#define W2_PAD 104          // halfs per W2T row      (208 B)
#define H_PAD  108          // floats per Hmid row    (432 B; rows m and m+8 land 32 banks apart)

typedef __attribute__((ext_vector_type(16))) _Float16 v16h;
typedef __attribute__((ext_vector_type(8)))  float    v8f;

union HFrag { v16h v; uint4 q[2]; _Float16 h[16]; };

// shared-memory byte budget
#define SMEM_BYTES (96*A_PAD*2 /*W1T*/ + 96*W2_PAD*2 /*W2T*/ + 96*4*2 /*b1,b2*/ \
                    + WAVES*(TILE*A_PAD*2 /*Ain*/ + TILE*H_PAD*4 /*Hmid*/ + 64 /*dst*/))

__device__ __forceinline__ void store_h8(_Float16* p, float4 a, float4 b) {
  union { _Float16 h[8]; uint4 q; } u;
  u.h[0] = (_Float16)a.x; u.h[1] = (_Float16)a.y;
  u.h[2] = (_Float16)a.z; u.h[3] = (_Float16)a.w;
  u.h[4] = (_Float16)b.x; u.h[5] = (_Float16)b.y;
  u.h[6] = (_Float16)b.z; u.h[7] = (_Float16)b.w;
  *(uint4*)p = u.q;               // ds_store_b128 (16B aligned by construction)
}

__device__ __forceinline__ v16h cvt16(float4 a, float4 b, float4 c, float4 d) {
  v16h r;
  r[0]=(_Float16)a.x; r[1]=(_Float16)a.y; r[2]=(_Float16)a.z; r[3]=(_Float16)a.w;
  r[4]=(_Float16)b.x; r[5]=(_Float16)b.y; r[6]=(_Float16)b.z; r[7]=(_Float16)b.w;
  r[8]=(_Float16)c.x; r[9]=(_Float16)c.y; r[10]=(_Float16)c.z; r[11]=(_Float16)c.w;
  r[12]=(_Float16)d.x; r[13]=(_Float16)d.y; r[14]=(_Float16)d.z; r[15]=(_Float16)d.w;
  return r;
}

// residual init: out_scalar = h_scalar, out_vector = h_vector
__global__ void slf_init_out(const float* __restrict__ hs,
                             const float* __restrict__ hv,
                             float* __restrict__ out) {
  int i = blockIdx.x * blockDim.x + threadIdx.x;
  if (i < N_NODES * SCALAR_DIM) out[i] = hs[i];
  if (i < N_NODES * VECTOR_DIM) out[(size_t)N_NODES * SCALAR_DIM + i] = hv[i];
}

__global__ __launch_bounds__(WAVES * 32)
void slf_edge_kernel(const float* __restrict__ hs,       // [N,64]
                     const float* __restrict__ hv,       // [N,32]
                     const int*   __restrict__ eidx,     // [2,E] (src row, dst row)
                     const float* __restrict__ pos,      // [N,2]
                     const float* __restrict__ orient,   // [N,1]
                     const float* __restrict__ W1,       // [163,96]
                     const float* __restrict__ b1,       // [96]
                     const float* __restrict__ W2,       // [96,96]
                     const float* __restrict__ b2,       // [96]
                     float*       __restrict__ out) {    // [N*64 | N*32]
  extern __shared__ unsigned char smem[];
  _Float16* W1T = (_Float16*)smem;                      // [96][A_PAD]  (W1T[n][k] = W1[k][n], f16)
  _Float16* W2T = W1T + 96 * A_PAD;                     // [96][W2_PAD]
  float*    b1s = (float*)(W2T + 96 * W2_PAD);          // [96]
  float*    b2s = b1s + 96;                             // [96]
  unsigned char* wbase = (unsigned char*)(b2s + 96);
  const int WB = TILE*A_PAD*2 + TILE*H_PAD*4 + 64;      // per-wave staging block (13376 B)

  const int tid  = threadIdx.x;
  const int lane = tid & 31;
  const int wave = tid >> 5;
  const int row  = lane & 15;    // WMMA M row / N column owned by this lane
  const int hh   = lane >> 4;    // lane-half selects K runs in A/B fragments

  _Float16* Ain  = (_Float16*)(wbase + wave * WB);               // [16][A_PAD] f16
  float*    Hmid = (float*)((unsigned char*)Ain + TILE*A_PAD*2); // [16][H_PAD] f32
  int*      dstW = (int*)((unsigned char*)Hmid + TILE*H_PAD*4);  // [16]

  // ---- stage weights (f16, transposed so B-fragment reads are contiguous) ----
  for (int idx = tid; idx < 96 * A_PAD; idx += blockDim.x) {
    int n = idx / A_PAD, k = idx - n * A_PAD;
    W1T[idx] = (k < IN_DIM) ? (_Float16)W1[k * HID + n] : (_Float16)0.f;
  }
  for (int idx = tid; idx < 96 * W2_PAD; idx += blockDim.x) {
    int n = idx / W2_PAD, k = idx - n * W2_PAD;
    W2T[idx] = (k < HID) ? (_Float16)W2[k * HID + n] : (_Float16)0.f;
  }
  if (tid < 96) { b1s[tid] = b1[tid]; b2s[tid] = b2[tid]; }
  __syncthreads();

  const int ntiles = (N_EDGES + TILE - 1) / TILE;
  const int gw = blockIdx.x * WAVES + wave;
  const int nw = gridDim.x * WAVES;

  for (int t = gw; t < ntiles; t += nw) {
    const int eBase = t * TILE;
    const int e  = eBase + row;
    const bool valid = (e < N_EDGES);
    const int ec = valid ? e : (N_EDGES - 1);
    const int src = eidx[ec];
    const int dst = eidx[N_EDGES + ec];
    const float alpha = orient[dst];
    _Float16* Arow = Ain + row * A_PAD;

    // ---- phase 1: build msg_input tile [16][192] in LDS (f16) ----
    if (hh == 0) {
      const float4* s4 = (const float4*)(hs + (size_t)src * SCALAR_DIM);
#pragma unroll
      for (int i = 0; i < 8; ++i) store_h8(Arow + 8 * i, s4[2*i], s4[2*i+1]);
      // rotated vector features 128..159: R(2*(beta-alpha)) applied to pairs
      const float beta = orient[src];
      float s, c;
      __sincosf(2.f * (beta - alpha), &s, &c);      // v_sin_f32 / v_cos_f32
      const float4* v4 = (const float4*)(hv + (size_t)src * VECTOR_DIM);
#pragma unroll
      for (int i = 0; i < 4; ++i) {
        float4 q0 = v4[2*i], q1 = v4[2*i+1], r0, r1;
        r0.x = c*q0.x - s*q0.y;  r0.y = s*q0.x + c*q0.y;
        r0.z = c*q0.z - s*q0.w;  r0.w = s*q0.z + c*q0.w;
        r1.x = c*q1.x - s*q1.y;  r1.y = s*q1.x + c*q1.y;
        r1.z = c*q1.z - s*q1.w;  r1.w = s*q1.z + c*q1.w;
        store_h8(Arow + 128 + 8 * i, r0, r1);
      }
      dstW[row] = valid ? dst : -1;
    } else {
      const float4* s4 = (const float4*)(hs + (size_t)dst * SCALAR_DIM);
#pragma unroll
      for (int i = 0; i < 8; ++i) store_h8(Arow + 64 + 8 * i, s4[2*i], s4[2*i+1]);
      // geo features 160..162 without atan2:
      //   cos(2phi) = (dx^2-dy^2)/r^2, sin(2phi) = 2*dx*dy/r^2  (fast v_rcp_f32)
      //   cos(2phi-2a) = cos2phi*c2a + sin2phi*s2a
      //   sin(2phi-2a) = sin2phi*c2a - cos2phi*s2a
      float2 ps = ((const float2*)pos)[src];
      float2 pd = ((const float2*)pos)[dst];
      float dx = ps.x - pd.x, dy = ps.y - pd.y;
      float r2 = fmaxf(dx*dx + dy*dy, 1e-30f);
      float inv = __builtin_amdgcn_rcpf(r2);        // v_rcp_f32
      float dist = sqrtf(r2) + 1e-6f;
      float c2p = (dx*dx - dy*dy) * inv;
      float s2p = (2.f * dx * dy) * inv;
      float s2a, c2a;
      __sincosf(2.f * alpha, &s2a, &c2a);
      float cg = c2p * c2a + s2p * s2a;
      float sg = s2p * c2a - c2p * s2a;
      union { _Float16 h[8]; uint4 q; } g;
      g.h[0] = (_Float16)dist; g.h[1] = (_Float16)cg; g.h[2] = (_Float16)sg;
      g.h[3] = (_Float16)0.f; g.h[4] = (_Float16)0.f; g.h[5] = (_Float16)0.f;
      g.h[6] = (_Float16)0.f; g.h[7] = (_Float16)0.f;
      *(uint4*)(Arow + 160) = g.q;
      uint4 z; z.x = z.y = z.z = z.w = 0u;
      *(uint4*)(Arow + 168) = z; *(uint4*)(Arow + 176) = z;
      *(uint4*)(Arow + 184) = z; *(uint4*)(Arow + 192) = z;
    }
    // LDS in-order within a wave -> A panel visible to the fragment loads below.

    // ---- phase 2: layer 1, 6 N-tiles x 6 K-chunks of v_wmma_f32_16x16x32_f16 ----
    HFrag A1[6];
#pragma unroll
    for (int kc = 0; kc < 6; ++kc) {
      const _Float16* p = Arow + kc * 32 + hh * 8;   // A layout: runs [h*8,+8) & [16+h*8,+8)
      A1[kc].q[0] = *(const uint4*)p;
      A1[kc].q[1] = *(const uint4*)(p + 16);
    }
#pragma unroll
    for (int nt = 0; nt < 6; ++nt) {
      const float bn = b1s[nt * 16 + row];
      v8f acc = {bn, bn, bn, bn, bn, bn, bn, bn};
      const _Float16* Wrow = W1T + (nt * 16 + row) * A_PAD + hh * 8;
#pragma unroll
      for (int kc = 0; kc < 6; ++kc) {
        HFrag B;
        B.q[0] = *(const uint4*)(Wrow + kc * 32);
        B.q[1] = *(const uint4*)(Wrow + kc * 32 + 16);
        acc = __builtin_amdgcn_wmma_f32_16x16x32_f16(false, A1[kc].v, false, B.v,
                                                     (short)0, acc, false, false);
      }
      // SiLU with fast v_rcp_f32; stage f32 (C/D layout: VGPR r holds M = 8*hh + r, N = row)
#pragma unroll
      for (int r = 0; r < 8; ++r) {
        float x = acc[r];
        float sg = __builtin_amdgcn_rcpf(1.f + __expf(-x));   // v_exp_f32 + v_rcp_f32
        Hmid[(8 * hh + r) * H_PAD + nt * 16 + row] = x * sg;
      }
    }

    // ---- phase 3: layer 2, 6 N-tiles x 3 K-chunks ----
    HFrag A2[3];
#pragma unroll
    for (int kc = 0; kc < 3; ++kc) {
      const float* p = Hmid + row * H_PAD + kc * 32 + hh * 8;
      A2[kc].v = cvt16(*(const float4*)p,        *(const float4*)(p + 4),
                       *(const float4*)(p + 16), *(const float4*)(p + 20));
    }
    int dloc[8];
#pragma unroll
    for (int r = 0; r < 8; ++r) dloc[r] = dstW[8 * hh + r];

#pragma unroll
    for (int nt = 0; nt < 6; ++nt) {
      const float bn = b2s[nt * 16 + row];
      v8f acc = {bn, bn, bn, bn, bn, bn, bn, bn};
      const _Float16* Wrow = W2T + (nt * 16 + row) * W2_PAD + hh * 8;
#pragma unroll
      for (int kc = 0; kc < 3; ++kc) {
        HFrag B;
        B.q[0] = *(const uint4*)(Wrow + kc * 32);
        B.q[1] = *(const uint4*)(Wrow + kc * 32 + 16);
        acc = __builtin_amdgcn_wmma_f32_16x16x32_f16(false, A2[kc].v, false, B.v,
                                                     (short)0, acc, false, false);
      }
      // ---- phase 4: segment-sum scatter (global_atomic_add_f32, no return) ----
      const int f = nt * 16 + row;          // output feature 0..95
#pragma unroll
      for (int r = 0; r < 8; ++r) {
        const int d = dloc[r];
        if (d >= 0) {
          if (nt < 4)   // f < 64 -> scalar part
            unsafeAtomicAdd(out + (size_t)d * SCALAR_DIM + f, acc[r]);
          else          // f >= 64 -> vector part
            unsafeAtomicAdd(out + (size_t)N_NODES * SCALAR_DIM
                                + (size_t)d * VECTOR_DIM + (f - SCALAR_DIM), acc[r]);
        }
      }
    }
  }
}

extern "C" void kernel_launch(void* const* d_in, const int* in_sizes, int n_in,
                              void* d_out, int out_size, void* d_ws, size_t ws_size,
                              hipStream_t stream) {
  const float* hs     = (const float*)d_in[0];
  const float* hv     = (const float*)d_in[1];
  const int*   eidx   = (const int*)d_in[2];   // harness: integer -> int32
  const float* pos    = (const float*)d_in[3];
  const float* orient = (const float*)d_in[4];
  const float* W1     = (const float*)d_in[5];
  const float* b1     = (const float*)d_in[6];
  const float* W2     = (const float*)d_in[7];
  const float* b2     = (const float*)d_in[8];
  float* out = (float*)d_out;

  (void)in_sizes; (void)n_in; (void)d_ws; (void)ws_size; (void)out_size;

  const int initN = N_NODES * SCALAR_DIM;  // covers both segments (vector is smaller)
  slf_init_out<<<(initN + 255) / 256, 256, 0, stream>>>(hs, hv, out);

  hipFuncSetAttribute((const void*)slf_edge_kernel,
                      hipFuncAttributeMaxDynamicSharedMemorySize, SMEM_BYTES);
  slf_edge_kernel<<<2000, WAVES * 32, SMEM_BYTES, stream>>>(
      hs, hv, eidx, pos, orient, W1, b1, W2, b2, out);
}
